// SelfAttention_74354473828731
// MI455X (gfx1250) — compile-verified, matrix operand order checked by reading, not run on hardware
//
#include <hip/hip_runtime.h>
#include <hip/hip_bf16.h>

// MI455X / gfx1250, wave32. Matrix math via v_wmma_f32_16x16x32_bf16.
// Tile staging via Tensor Data Mover (tensor_load_to_lds, TENSORcnt),
// double-buffered so DMA overlaps WMMA compute.

typedef __attribute__((ext_vector_type(16))) __bf16 v16bf;
typedef __attribute__((ext_vector_type(8)))  float  v8f;

#define DIM      512
#define SEQ      4096
#define BATCH    2
#define NHEADS   8
#define HEADDIM  64
#define MROWS    (BATCH * SEQ)      // 8192

__device__ __forceinline__ unsigned short f2bf(float f) {
    union { float f; unsigned int u; } c; c.f = f;
    unsigned int r = c.u + 0x7FFFu + ((c.u >> 16) & 1u);   // round to nearest even
    return (unsigned short)(r >> 16);
}

// 16x32 bf16 WMMA fragment (A- or B-side): lane = 16*h + idx, elements 0..7 at
// K=8h..8h+7, elements 8..15 at K=16+8h..; `row` points at row_start+kstep*32.
__device__ __forceinline__ v16bf load_frag(const unsigned short* row, int h) {
    union { uint4 q[2]; v16bf v; } u;
    u.q[0] = *(const uint4*)(row + 8 * h);
    u.q[1] = *(const uint4*)(row + 16 + 8 * h);
    return u.v;
}

// ---------------------------------------------------------------- TDM load
// 2-D tile load global->LDS. Rows of tileQw 8-byte elements, tileRows rows,
// row stride strideQw (8-byte units). D# per CDNA5 ISA.
typedef __attribute__((ext_vector_type(4))) unsigned tdm_g0_t;
typedef __attribute__((ext_vector_type(8))) unsigned tdm_g1_t;

__device__ __forceinline__ void tdm_load_2d(unsigned ldsOff, const void* gsrc,
                                            unsigned tileQw, unsigned tileRows,
                                            unsigned strideQw, unsigned tensorRows) {
    unsigned long long ga = (unsigned long long)(size_t)gsrc;
    tdm_g0_t g0;
    g0[0] = 1u;                                            // count=1 (valid user D#)
    g0[1] = ldsOff;                                        // lds_addr (bytes)
    g0[2] = (unsigned)ga;                                  // global_addr[31:0]
    g0[3] = ((unsigned)(ga >> 32) & 0x01FFFFFFu) | (2u << 30); // [56:32] | type=2
    tdm_g1_t g1;
    g1[0] = 3u << 16;                                      // data_size=3 (8 bytes)
    g1[1] = (strideQw & 0xFFFFu) << 16;                    // tensor_dim0[15:0]
    g1[2] = (strideQw >> 16) | ((tensorRows & 0xFFFFu) << 16); // dim0 hi | dim1 lo
    g1[3] = (tensorRows >> 16) | (tileQw << 16);           // dim1 hi | tile_dim0
    g1[4] = tileRows;                                      // tile_dim1
    g1[5] = strideQw;                                      // tensor_dim0_stride lo
    g1[6] = 0u;
    g1[7] = 0u;
    asm volatile("tensor_load_to_lds %0, %1" :: "s"(g0), "s"(g1) : "memory");
}

__device__ __forceinline__ void tdm_wait() {
#if __has_builtin(__builtin_amdgcn_s_wait_tensorcnt)
    __builtin_amdgcn_s_wait_tensorcnt(0);
#else
    asm volatile("s_wait_tensorcnt 0x0" ::: "memory");
#endif
}

// ---------------------------------------------------------------- convert x
__global__ __launch_bounds__(256) void cvt_bf16_kernel(
    const float* __restrict__ in, unsigned short* __restrict__ out, int n) {
    int i = (blockIdx.x * 256 + threadIdx.x) * 4;
    if (i + 3 < n) {
        float4 v = *(const float4*)(in + i);
        unsigned int lo = (unsigned int)f2bf(v.x) | ((unsigned int)f2bf(v.y) << 16);
        unsigned int hi = (unsigned int)f2bf(v.z) | ((unsigned int)f2bf(v.w) << 16);
        *(uint2*)(out + i) = make_uint2(lo, hi);
    }
}

// ------------------------------------------- convert + transpose 512x512 W
__global__ __launch_bounds__(256) void cvt_transpose_kernel(
    const float* __restrict__ W, unsigned short* __restrict__ Wt) {
    int id = blockIdx.x * 256 + threadIdx.x;
    int k = id >> 9, n = id & 511;
    Wt[n * 512 + k] = f2bf(W[id]);
}

// ------------------------------------------------------------------- GEMM
// C[M,512] = A[M,512](bf16) @ Wt^T + bias.  Wt is [n][k] bf16.
// mode 0: scatter bf16 into [B,H,S,Hd].  mode 1: fp32 row-major + bias.
// Double-buffered TDM staging: TDM for tile i+1 overlaps WMMA on tile i.
#define GM_BM 128
#define GM_BN 64
#define GM_BK 64
#define GM_ASZ (GM_BM * GM_BK)
#define GM_BSZ (GM_BN * GM_BK)
__global__ __launch_bounds__(256) void gemm_bf16_kernel(
    const unsigned short* __restrict__ A,
    const unsigned short* __restrict__ Bt,
    const float* __restrict__ bias,
    void* __restrict__ Out, int mode) {
    __shared__ __align__(16) unsigned short As[2 * GM_ASZ];
    __shared__ __align__(16) unsigned short Bs[2 * GM_BSZ];
    const int K = DIM;
    int tid  = threadIdx.x;
    int bm   = blockIdx.x, bn = blockIdx.y;
    int wave = tid >> 5, lane = tid & 31;
    int h = lane >> 4, nl = lane & 15;
    int wm = (wave >> 1) * 32, wn = (wave & 1) * 32;

    const unsigned short* Ab0 = A  + (size_t)bm * GM_BM * K;
    const unsigned short* Bb0 = Bt + (size_t)bn * GM_BN * K;

    v8f acc[2][2] = {};
    int buf = 0;

    if (wave == 0) {                       // prologue: stage tile 0
        tdm_load_2d((unsigned)(size_t)(void*)As, Ab0, GM_BK / 4, GM_BM, K / 4, MROWS);
        tdm_load_2d((unsigned)(size_t)(void*)Bs, Bb0, GM_BK / 4, GM_BN, K / 4, DIM);
    }

    for (int k0 = 0; k0 < K; k0 += GM_BK) {
        if (wave == 0) tdm_wait();         // tile i landed
        __syncthreads();                   // publish tile i; prev buf free
        if (wave == 0 && k0 + GM_BK < K) { // prefetch tile i+1 into other buf
            tdm_load_2d((unsigned)(size_t)(void*)(As + (buf ^ 1) * GM_ASZ),
                        Ab0 + k0 + GM_BK, GM_BK / 4, GM_BM, K / 4, MROWS);
            tdm_load_2d((unsigned)(size_t)(void*)(Bs + (buf ^ 1) * GM_BSZ),
                        Bb0 + k0 + GM_BK, GM_BK / 4, GM_BN, K / 4, DIM);
        }
        const unsigned short* as = As + buf * GM_ASZ;
        const unsigned short* bs = Bs + buf * GM_BSZ;
#pragma unroll
        for (int ks = 0; ks < 2; ++ks) {
            v16bf af[2], wf[2];
            af[0] = load_frag(as + (wm + nl) * GM_BK + ks * 32, h);
            af[1] = load_frag(as + (wm + 16 + nl) * GM_BK + ks * 32, h);
            wf[0] = load_frag(bs + (wn + nl) * GM_BK + ks * 32, h);
            wf[1] = load_frag(bs + (wn + 16 + nl) * GM_BK + ks * 32, h);
#pragma unroll
            for (int mi = 0; mi < 2; ++mi)
#pragma unroll
                for (int ni = 0; ni < 2; ++ni)
                    acc[mi][ni] = __builtin_amdgcn_wmma_f32_16x16x32_bf16(
                        false, af[mi], false, wf[ni], (short)0, acc[mi][ni],
                        false, false);
        }
        buf ^= 1;
    }

#pragma unroll
    for (int mi = 0; mi < 2; ++mi)
#pragma unroll
        for (int ni = 0; ni < 2; ++ni) {
            int nCol = bn * GM_BN + wn + ni * 16 + nl;
            float bv = bias[nCol];
#pragma unroll
            for (int r = 0; r < 8; ++r) {
                int mRow = bm * GM_BM + wm + mi * 16 + r + 8 * h;
                float v = acc[mi][ni][r] + bv;
                if (mode == 0) {
                    int b = mRow >> 12, s = mRow & 4095;
                    int hh = nCol >> 6, hd = nCol & 63;
                    ((unsigned short*)Out)[((((size_t)(b * NHEADS + hh)) * SEQ + s) << 6) + hd] = f2bf(v);
                } else {
                    ((float*)Out)[(size_t)mRow * DIM + nCol] = v;
                }
            }
        }
}

// scatter 8 consecutive bf16 (one uint4) into transposed LDS V tile
__device__ __forceinline__ void store8T(unsigned short* Vt, int col, int key, uint4 v) {
    unsigned short* p = Vt + col * 64 + key;
    p[0 * 64] = (unsigned short)v.x; p[1 * 64] = (unsigned short)(v.x >> 16);
    p[2 * 64] = (unsigned short)v.y; p[3 * 64] = (unsigned short)(v.y >> 16);
    p[4 * 64] = (unsigned short)v.z; p[5 * 64] = (unsigned short)(v.z >> 16);
    p[6 * 64] = (unsigned short)v.w; p[7 * 64] = (unsigned short)(v.w >> 16);
}

// --------------------------------------------------------- flash attention
// grid = (SEQ/128, B*H); each wave owns 16 query rows, block stages 64 keys.
// K tile: double-buffered TDM.  V tile: next-tile global loads issued at loop
// top, transposed LDS scatter after compute (stores published by next barrier).
#define AT_TSZ (64 * 64)
__global__ __launch_bounds__(256) void attn_kernel(
    const unsigned short* __restrict__ Q,
    const unsigned short* __restrict__ Kt,
    const unsigned short* __restrict__ V,
    unsigned short* __restrict__ Oa) {
    __shared__ __align__(16) unsigned short Ks[2 * AT_TSZ];   // [key][hd]
    __shared__ __align__(16) unsigned short Vt[2 * AT_TSZ];   // [hd][key]
    __shared__ __align__(16) unsigned short Pls[8 * 16 * 64]; // per-wave P patch

    const float scale = 0.04419417382415922f;                 // 1/sqrt(512)
    int tid = threadIdx.x, wave = tid >> 5, lane = tid & 31;
    int h = lane >> 4, nl = lane & 15;
    int bh = blockIdx.y;
    int b = bh >> 3, hh = bh & 7;
    size_t seqBase = (size_t)bh * SEQ;
    int qs = blockIdx.x * 128 + wave * 16;
    int vkey = tid >> 3, vcol = (tid & 7) * 8;

    v16bf qf[2];
    {
        const unsigned short* qrow = Q + (seqBase + qs + nl) * HEADDIM;
        qf[0] = load_frag(qrow, h);
        qf[1] = load_frag(qrow + 32, h);
    }

    v8f oacc[4] = {};
    float mOld[8], lOld[8];
#pragma unroll
    for (int r = 0; r < 8; ++r) { mOld[r] = -3.0e38f; lOld[r] = 0.f; }

    int buf = 0;
    {   // prologue: stage tile 0
        const unsigned short* s0 = V + (seqBase + vkey) * HEADDIM + vcol;
        uint4 va = *(const uint4*)s0;
        uint4 vb = *(const uint4*)(s0 + 32 * HEADDIM);
        store8T(Vt, vcol, vkey, va);
        store8T(Vt, vcol, vkey + 32, vb);
        if (wave == 0)
            tdm_load_2d((unsigned)(size_t)(void*)Ks, Kt + seqBase * HEADDIM,
                        16, 64, 16, SEQ);
    }

    for (int kb = 0; kb < SEQ; kb += 64) {
        bool next = (kb + 64) < SEQ;
        uint4 va, vb;
        if (next) {                         // global loads for V tile i+1 now;
            const unsigned short* s0 =      // the whole compute hides them
                V + (seqBase + kb + 64 + vkey) * HEADDIM + vcol;
            va = *(const uint4*)s0;
            vb = *(const uint4*)(s0 + 32 * HEADDIM);
        }
        if (wave == 0) tdm_wait();          // K tile i landed
        __syncthreads();                    // publish tile i; buf^1 free
        if (next && wave == 0)              // prefetch K tile i+1
            tdm_load_2d((unsigned)(size_t)(void*)(Ks + (buf ^ 1) * AT_TSZ),
                        Kt + (seqBase + kb + 64) * HEADDIM, 16, 64, 16, SEQ);

        const unsigned short* ks_ = Ks + buf * AT_TSZ;
        const unsigned short* vt_ = Vt + buf * AT_TSZ;

        // S = Q @ K^T   (16 x 64, four 16x16 accum tiles)
        v8f sacc[4] = {};
#pragma unroll
        for (int ks = 0; ks < 2; ++ks) {
            v16bf kf[4];
#pragma unroll
            for (int t = 0; t < 4; ++t)
                kf[t] = load_frag(ks_ + (t * 16 + nl) * 64 + ks * 32, h);
#pragma unroll
            for (int t = 0; t < 4; ++t)
                sacc[t] = __builtin_amdgcn_wmma_f32_16x16x32_bf16(
                    false, qf[ks], false, kf[t], (short)0, sacc[t], false, false);
        }

        // online softmax per row (rows r+8h; 16-lane halves hold 16 key cols)
#pragma unroll
        for (int r = 0; r < 8; ++r) {
            float mx = fmaxf(fmaxf(sacc[0][r], sacc[1][r]),
                             fmaxf(sacc[2][r], sacc[3][r]));
            mx = fmaxf(mx, __shfl_xor(mx, 1, 32));
            mx = fmaxf(mx, __shfl_xor(mx, 2, 32));
            mx = fmaxf(mx, __shfl_xor(mx, 4, 32));
            mx = fmaxf(mx, __shfl_xor(mx, 8, 32));
            mx *= scale;
            float mNew = fmaxf(mOld[r], mx);
            float alpha = __expf(mOld[r] - mNew);
            float rs = 0.f;
#pragma unroll
            for (int t = 0; t < 4; ++t) {
                float p = __expf(sacc[t][r] * scale - mNew);
                sacc[t][r] = p;
                rs += p;
            }
            rs += __shfl_xor(rs, 1, 32);
            rs += __shfl_xor(rs, 2, 32);
            rs += __shfl_xor(rs, 4, 32);
            rs += __shfl_xor(rs, 8, 32);
            lOld[r] = lOld[r] * alpha + rs;
            mOld[r] = mNew;
#pragma unroll
            for (int t = 0; t < 4; ++t) oacc[t][r] *= alpha;
        }

        // C-layout P -> A-layout via wave-private LDS patch
        unsigned short* pw = Pls + wave * 16 * 64;
#pragma unroll
        for (int r = 0; r < 8; ++r) {
            int m = r + 8 * h;
#pragma unroll
            for (int t = 0; t < 4; ++t)
                pw[m * 64 + t * 16 + nl] = f2bf(sacc[t][r]);
        }

        // O += P @ V   (contraction over 64 keys)
#pragma unroll
        for (int ks = 0; ks < 2; ++ks) {
            v16bf pf = load_frag(pw + nl * 64 + ks * 32, h);
            v16bf vf[4];
#pragma unroll
            for (int t = 0; t < 4; ++t)
                vf[t] = load_frag(vt_ + (t * 16 + nl) * 64 + ks * 32, h);
#pragma unroll
            for (int t = 0; t < 4; ++t)
                oacc[t] = __builtin_amdgcn_wmma_f32_16x16x32_bf16(
                    false, pf, false, vf[t], (short)0, oacc[t], false, false);
        }

        if (next) {                          // scatter V tile i+1 (published at
            unsigned short* vdst = Vt + (buf ^ 1) * AT_TSZ;  // next barrier)
            store8T(vdst, vcol, vkey, va);
            store8T(vdst, vcol, vkey + 32, vb);
        }
        buf ^= 1;
    }

    // normalize + write bf16 rows into [B*S, D] for the output projection
#pragma unroll
    for (int r = 0; r < 8; ++r) {
        int m = r + 8 * h;
        float inv = 1.0f / lOld[r];
        size_t row = (size_t)b * SEQ + qs + m;
#pragma unroll
        for (int t = 0; t < 4; ++t)
            Oa[row * DIM + hh * HEADDIM + t * 16 + nl] = f2bf(oacc[t][r] * inv);
    }
}

// ------------------------------------------------------------------ launch
extern "C" void kernel_launch(void* const* d_in, const int* in_sizes, int n_in,
                              void* d_out, int out_size, void* d_ws, size_t ws_size,
                              hipStream_t stream) {
    const float* x  = (const float*)d_in[0];
    const float* Wq = (const float*)d_in[1];
    const float* bq = (const float*)d_in[2];
    const float* Wk = (const float*)d_in[3];
    const float* bk = (const float*)d_in[4];
    const float* Wv = (const float*)d_in[5];
    const float* bv = (const float*)d_in[6];
    const float* Wo = (const float*)d_in[7];
    const float* bo = (const float*)d_in[8];

    const size_t NX = (size_t)MROWS * DIM;   // 8192*512
    const size_t NW = (size_t)DIM * DIM;     // 512*512

    unsigned short* xb  = (unsigned short*)d_ws;
    unsigned short* WqT = xb  + NX;
    unsigned short* WkT = WqT + NW;
    unsigned short* WvT = WkT + NW;
    unsigned short* WoT = WvT + NW;
    unsigned short* Qb  = WoT + NW;
    unsigned short* Kb  = Qb + NX;
    unsigned short* Vb  = Kb + NX;
    unsigned short* Ab  = Vb + NX;          // total ~44 MB of d_ws

    cvt_bf16_kernel<<<(int)(NX / 4 / 256), 256, 0, stream>>>(x, xb, (int)NX);
    cvt_transpose_kernel<<<(int)(NW / 256), 256, 0, stream>>>(Wq, WqT);
    cvt_transpose_kernel<<<(int)(NW / 256), 256, 0, stream>>>(Wk, WkT);
    cvt_transpose_kernel<<<(int)(NW / 256), 256, 0, stream>>>(Wv, WvT);
    cvt_transpose_kernel<<<(int)(NW / 256), 256, 0, stream>>>(Wo, WoT);

    dim3 gg(MROWS / GM_BM, DIM / GM_BN);             // 64 x 8
    gemm_bf16_kernel<<<gg, 256, 0, stream>>>(xb, WqT, bq, (void*)Qb, 0);
    gemm_bf16_kernel<<<gg, 256, 0, stream>>>(xb, WkT, bk, (void*)Kb, 0);
    gemm_bf16_kernel<<<gg, 256, 0, stream>>>(xb, WvT, bv, (void*)Vb, 0);

    attn_kernel<<<dim3(SEQ / 128, BATCH * NHEADS), 256, 0, stream>>>(Qb, Kb, Vb, Ab);

    gemm_bf16_kernel<<<gg, 256, 0, stream>>>(Ab, WoT, bo, d_out, 1);
}
